// Decoder_62234076119175
// MI455X (gfx1250) — matrix operand order repaired and checked by reading, not compiled, not run on hardware
//
#include <hip/hip_runtime.h>
#include <hip/hip_bf16.h>

typedef unsigned short u16;
typedef unsigned int   u32;
typedef __attribute__((ext_vector_type(16))) __bf16 v16bf;
typedef __attribute__((ext_vector_type(8)))  float  v8f;
typedef __attribute__((ext_vector_type(4)))  u32    uint4v;

union Frag { uint4v u[2]; v16bf v; };

#define NEGF (-1.0e9f)
#define SEQ 1024
#define DH 128

__device__ __forceinline__ u16 f2b(float f) {
    u32 u = __float_as_uint(f);
    u += 0x7FFFu + ((u >> 16) & 1u);
    return (u16)(u >> 16);
}
__device__ __forceinline__ float b2f(u16 h) {
    return __uint_as_float(((u32)h) << 16);
}

// ---------------------------------------------------------------------------
// Elementwise fp32 -> bf16 convert (row-major preserved)
// ---------------------------------------------------------------------------
__global__ void convert_f2b_kernel(const float* __restrict__ in, u16* __restrict__ out, int total) {
    int i = blockIdx.x * blockDim.x + threadIdx.x;
    if (i < total) out[i] = f2b(in[i]);
}

// ---------------------------------------------------------------------------
// Transpose + convert: in [K,N] fp32 row-major -> out [N,K] bf16
// ---------------------------------------------------------------------------
__global__ void transpose_f2b_kernel(const float* __restrict__ in, u16* __restrict__ out,
                                     int K, int N) {
    int i = blockIdx.x * blockDim.x + threadIdx.x;
    if (i >= K * N) return;
    int n = i % N;
    int k = i / N;
    out[(size_t)n * K + k] = f2b(in[i]);
}

// ---------------------------------------------------------------------------
// WMMA GEMM: C[M,N] f32 = A[M,K] bf16 (row-major) x Bt[N,K] bf16.
// 256 threads = 8 waves; each wave computes a 16 x (16*NT) tile, sharing one
// A fragment across NT B fragments (2.5 loads/WMMA at NT=4 vs 4 at NT=1).
// ---------------------------------------------------------------------------
template <int NT>
__global__ __launch_bounds__(256) void gemm_bf16_kernel(const u16* __restrict__ A,
                                                        const u16* __restrict__ Bt,
                                                        float* __restrict__ C,
                                                        int M, int N, int K) {
    int lane = threadIdx.x & 31;
    int w    = threadIdx.x >> 5;
    int tiles_n = N / (16 * NT);
    int tiles   = (M >> 4) * tiles_n;
    int tid = blockIdx.x * 8 + w;
    if (tid >= tiles) return;
    int tm  = tid / tiles_n;
    int tn0 = (tid % tiles_n) * NT;
    int mrow = lane & 15;
    int hi   = lane >> 4;

    const u16* ap0 = A  + (size_t)(tm * 16 + mrow) * K;
    const u16* bp0 = Bt + (size_t)(tn0 * 16 + mrow) * K;
    v8f acc[NT];
    #pragma unroll
    for (int i = 0; i < NT; ++i) acc[i] = (v8f){};

    for (int k0 = 0; k0 < K; k0 += 32) {
        Frag a;
        a.u[0] = *(const uint4v*)(ap0 + k0 + hi * 16);
        a.u[1] = *(const uint4v*)(ap0 + k0 + hi * 16 + 8);
        #pragma unroll
        for (int i = 0; i < NT; ++i) {
            Frag b;
            const u16* bp = bp0 + (size_t)i * 16 * K + k0 + hi * 16;
            b.u[0] = *(const uint4v*)(bp);
            b.u[1] = *(const uint4v*)(bp + 8);
            acc[i] = __builtin_amdgcn_wmma_f32_16x16x32_bf16(false, a.v, false, b.v,
                                                             (short)0, acc[i], false, false);
        }
    }
    #pragma unroll
    for (int i = 0; i < NT; ++i) {
        float* cp = C + (size_t)(tm * 16 + hi * 8) * N + (tn0 + i) * 16 + mrow;
        #pragma unroll
        for (int r = 0; r < 8; ++r) cp[(size_t)r * N] = acc[i][r];
    }
}

// ---------------------------------------------------------------------------
// RoPE + layout change: x[S, nheads*128] f32 -> out[nheads][S][128] bf16
// ---------------------------------------------------------------------------
__global__ void rope_kernel(const float* __restrict__ x, u16* __restrict__ out, int nheads) {
    int idx = blockIdx.x * blockDim.x + threadIdx.x;
    int total = SEQ * nheads * DH;
    if (idx >= total) return;
    int d  = idx & 127;
    int t  = idx >> 7;
    int hh = t % nheads;
    int s  = t / nheads;
    const float* row = x + (size_t)s * nheads * DH + hh * DH;
    int fr = d & 63;
    // inv_freq = 10000^(-fr/64) = exp(-fr * ln(10000)/64)
    float ang = (float)s * __expf(-(float)fr * 0.14391156514261515f);
    float c = __cosf(ang), sn = __sinf(ang);
    float v;
    if (d < 64) v = row[d] * c - row[d + 64] * sn;
    else        v = row[d] * c + row[d - 64] * sn;
    out[((size_t)hh * SEQ + s) * DH + d] = f2b(v);
}

// ---------------------------------------------------------------------------
// Low-rank positional mix: out[h][s][d] = sum_r lr[s][r] * pos[s][h][r][d]
// lr [S,32] f32, pos [S,H,32,128] f32 -> out [H][S][128] bf16
// ---------------------------------------------------------------------------
__global__ void mixpos_kernel(const float* __restrict__ lr, const float* __restrict__ pos,
                              u16* __restrict__ out) {
    int idx = blockIdx.x * blockDim.x + threadIdx.x;
    if (idx >= 32 * SEQ * DH) return;
    int d = idx & 127;
    int t = idx >> 7;
    int h = t & 31;
    int s = t >> 5;
    const float* pr = pos + (((size_t)s * 32 + h) * 32) * DH + d;
    const float* lp = lr + (size_t)s * 32;
    float acc = 0.0f;
    #pragma unroll
    for (int r = 0; r < 32; ++r) acc += lp[r] * pr[(size_t)r * DH];
    out[((size_t)h * SEQ + s) * DH + d] = f2b(acc);
}

// ---------------------------------------------------------------------------
// Fused sparse attention per (head, 16-query tile):
//  Phase A: true & draft scores via WMMA -> LDS (bf16)
//  Phase B: per-row top-512 select (binary search on sortable uints) + softmax
//  Phase C: probs @ V via WMMA -> attn [S][H*D] bf16
// ---------------------------------------------------------------------------
__global__ __launch_bounds__(256) void attn_kernel(const u16* __restrict__ qr,  // [32][S][128]
                                                   const u16* __restrict__ kr,  // [8][S][128]
                                                   const u16* __restrict__ pq,  // [32][S][128]
                                                   const u16* __restrict__ pk,  // [32][S][128]
                                                   const u16* __restrict__ vt,  // [8][128][S]
                                                   u16* __restrict__ attn) {    // [S][4096]
    __shared__ alignas(16) u16 s_true[16][SEQ];   // 32 KB
    __shared__ alignas(16) u16 s_draft[16][SEQ];  // 32 KB (reused for probs bf16)

    int qt = blockIdx.x;        // query tile 0..63
    int h  = blockIdx.y;        // head 0..31
    int kvh = h >> 2;           // GQA: 4 heads per kv head
    int lane = threadIdx.x & 31;
    int w    = threadIdx.x >> 5;
    int mrow = lane & 15;
    int hi   = lane >> 4;
    const float scale = 0.08838834764831845f;  // 1/sqrt(128)

    // Preload A fragments (query rows) for the 4 K-steps over D=128.
    Frag aq[4], ad[4];
    {
        const u16* qp = qr + ((size_t)h * SEQ + qt * 16 + mrow) * DH;
        const u16* pp = pq + ((size_t)h * SEQ + qt * 16 + mrow) * DH;
        #pragma unroll
        for (int ks = 0; ks < 4; ++ks) {
            const u16* q1 = qp + ks * 32;
            aq[ks].u[0] = *(const uint4v*)(q1 + hi * 16);
            aq[ks].u[1] = *(const uint4v*)(q1 + hi * 16 + 8);
            const u16* p1 = pp + ks * 32;
            ad[ks].u[0] = *(const uint4v*)(p1 + hi * 16);
            ad[ks].u[1] = *(const uint4v*)(p1 + hi * 16 + 8);
        }
    }

    // Phase A: each wave covers key tiles kt = it*8 + w.
    for (int it = 0; it < 8; ++it) {
        int kt = it * 8 + w;
        v8f at = {};
        v8f ad_acc = {};
        const u16* kp  = kr + ((size_t)kvh * SEQ + kt * 16 + mrow) * DH;
        const u16* pkp = pk + ((size_t)h   * SEQ + kt * 16 + mrow) * DH;
        #pragma unroll
        for (int ks = 0; ks < 4; ++ks) {
            Frag b1, b2;
            const u16* p1 = kp + ks * 32 + hi * 16;
            b1.u[0] = *(const uint4v*)(p1);
            b1.u[1] = *(const uint4v*)(p1 + 8);
            at = __builtin_amdgcn_wmma_f32_16x16x32_bf16(false, aq[ks].v, false, b1.v,
                                                         (short)0, at, false, false);
            const u16* p2 = pkp + ks * 32 + hi * 16;
            b2.u[0] = *(const uint4v*)(p2);
            b2.u[1] = *(const uint4v*)(p2 + 8);
            ad_acc = __builtin_amdgcn_wmma_f32_16x16x32_bf16(false, ad[ks].v, false, b2.v,
                                                             (short)0, ad_acc, false, false);
        }
        #pragma unroll
        for (int r = 0; r < 8; ++r) {
            int row = r + hi * 8;
            s_true[row][kt * 16 + mrow]  = f2b(at[r] * scale);
            s_draft[row][kt * 16 + mrow] = f2b(ad_acc[r]);
        }
    }
    __syncthreads();

    // Phase B: each wave handles rows w and w+8.
    for (int rr = 0; rr < 2; ++rr) {
        int row = w + rr * 8;
        int qglob = qt * 16 + row;
        u32 keys[32];
        #pragma unroll
        for (int j = 0; j < 32; ++j) {
            int col = lane + j * 32;
            float v = b2f(s_draft[row][col]);
            if (col > qglob) v = NEGF;       // causal mask on draft score
            u32 u = __float_as_uint(v);
            keys[j] = (u & 0x80000000u) ? ~u : (u | 0x80000000u);
        }
        // Binary search for the 512th-largest key: max t with count(>=t) >= 512.
        u32 t = 0u;
        for (int b = 31; b >= 0; --b) {
            u32 cand = t | (1u << b);
            int cnt = 0;
            #pragma unroll
            for (int j = 0; j < 32; ++j) cnt += (keys[j] >= cand) ? 1 : 0;
            for (int off = 16; off; off >>= 1) cnt += __shfl_xor(cnt, off, 32);
            if (cnt >= 512) t = cand;
        }
        // Masked softmax over true score.
        float logit[32];
        float mval = -3.0e38f;
        #pragma unroll
        for (int j = 0; j < 32; ++j) {
            int col = lane + j * 32;
            bool ok = (col <= qglob) && (keys[j] >= t);
            float v = ok ? b2f(s_true[row][col]) : NEGF;
            logit[j] = v;
            mval = fmaxf(mval, v);
        }
        for (int off = 16; off; off >>= 1) mval = fmaxf(mval, __shfl_xor(mval, off, 32));
        float sum = 0.0f;
        #pragma unroll
        for (int j = 0; j < 32; ++j) {
            float e = __expf(logit[j] - mval);
            logit[j] = e;
            sum += e;
        }
        for (int off = 16; off; off >>= 1) sum += __shfl_xor(sum, off, 32);
        float inv = 1.0f / sum;
        #pragma unroll
        for (int j = 0; j < 32; ++j) {
            int col = lane + j * 32;
            s_draft[row][col] = f2b(logit[j] * inv);  // reuse draft LDS as probs (own row only)
        }
    }
    __syncthreads();

    // Phase C: probs[16,1024] @ V -> wave w computes d-tile w (16 cols of D).
    v8f acc = {};
    int d0 = w * 16;
    const u16* vp = vt + ((size_t)kvh * DH + d0 + mrow) * SEQ;
    for (int kt = 0; kt < 32; ++kt) {
        Frag a, b;
        const u16* ap = &s_draft[mrow][kt * 32];
        a.u[0] = *(const uint4v*)(ap + hi * 16);
        a.u[1] = *(const uint4v*)(ap + hi * 16 + 8);
        const u16* bp = vp + kt * 32 + hi * 16;
        b.u[0] = *(const uint4v*)(bp);
        b.u[1] = *(const uint4v*)(bp + 8);
        acc = __builtin_amdgcn_wmma_f32_16x16x32_bf16(false, a.v, false, b.v,
                                                      (short)0, acc, false, false);
    }
    u16* op = attn + (size_t)(qt * 16 + hi * 8) * 4096 + h * DH + d0 + mrow;
    #pragma unroll
    for (int r = 0; r < 8; ++r) op[(size_t)r * 4096] = f2b(acc[r]);
}

// ---------------------------------------------------------------------------
// Host-side launch
// ---------------------------------------------------------------------------
extern "C" void kernel_launch(void* const* d_in, const int* in_sizes, int n_in,
                              void* d_out, int out_size, void* d_ws, size_t ws_size,
                              hipStream_t stream) {
    (void)in_sizes; (void)n_in; (void)out_size; (void)ws_size;
    const float* hs    = (const float*)d_in[0];  // [1024,4096]
    const float* wq    = (const float*)d_in[1];  // [4096,4096]
    const float* wk    = (const float*)d_in[2];  // [4096,1024]
    const float* wv    = (const float*)d_in[3];  // [4096,1024]
    const float* wo    = (const float*)d_in[4];  // [4096,4096]
    const float* mixq  = (const float*)d_in[5];  // [4096,32]
    const float* mixk  = (const float*)d_in[6];  // [4096,32]
    const float* q_pos = (const float*)d_in[7];  // [1024,32,32,128]
    const float* k_pos = (const float*)d_in[8];  // [1024,32,32,128]
    float* out = (float*)d_out;                  // [1024,4096]

    char* ws = (char*)d_ws;
    size_t off = 0;
    auto alloc = [&](size_t bytes) -> void* {
        void* p = ws + off;
        off += (bytes + 255) & ~(size_t)255;
        return p;
    };
    const size_t E = 4096, S = 1024, HD = 4096, KVD = 1024;
    u16* hsb   = (u16*)alloc(S * E * 2);        // hidden bf16 [S][E]
    u16* wqt   = (u16*)alloc(HD * E * 2);       // [HD][E]
    u16* wkt   = (u16*)alloc(KVD * E * 2);      // [KVD][E]
    u16* wvt   = (u16*)alloc(KVD * E * 2);
    u16* wot   = (u16*)alloc(E * HD * 2);       // [E][HD]
    u16* mixqt = (u16*)alloc(32 * E * 2);       // [32][E]
    u16* mixkt = (u16*)alloc(32 * E * 2);
    float* qf  = (float*)alloc(S * HD * 4);     // [S][HD]
    float* kf  = (float*)alloc(S * KVD * 4);    // [S][KVD]
    float* vf  = (float*)alloc(S * KVD * 4);
    float* lrq = (float*)alloc(S * 32 * 4);
    float* lrk = (float*)alloc(S * 32 * 4);
    u16* qrb   = (u16*)alloc(32 * S * 128 * 2); // [H][S][D]
    u16* krb   = (u16*)alloc(8 * S * 128 * 2);  // [KVH][S][D]
    u16* vtb   = (u16*)alloc(8 * 128 * S * 2);  // [KVH][D][S]
    u16* pqb   = (u16*)alloc(32 * S * 128 * 2); // [H][S][D]
    u16* pkb   = (u16*)alloc(32 * S * 128 * 2);
    u16* attnb = (u16*)alloc(S * HD * 2);       // [S][H*D]

    const int TB = 256;
    // 1) hidden -> bf16
    convert_f2b_kernel<<<(int)((S * E + TB - 1) / TB), TB, 0, stream>>>(hs, hsb, (int)(S * E));
    // 2) weight transposes -> bf16 [N][K]
    transpose_f2b_kernel<<<(int)((E * HD + TB - 1) / TB), TB, 0, stream>>>(wq, wqt, (int)E, (int)HD);
    transpose_f2b_kernel<<<(int)((E * KVD + TB - 1) / TB), TB, 0, stream>>>(wk, wkt, (int)E, (int)KVD);
    transpose_f2b_kernel<<<(int)((E * KVD + TB - 1) / TB), TB, 0, stream>>>(wv, wvt, (int)E, (int)KVD);
    transpose_f2b_kernel<<<(int)((HD * E + TB - 1) / TB), TB, 0, stream>>>(wo, wot, (int)HD, (int)E);
    transpose_f2b_kernel<<<(int)((E * 32 + TB - 1) / TB), TB, 0, stream>>>(mixq, mixqt, (int)E, 32);
    transpose_f2b_kernel<<<(int)((E * 32 + TB - 1) / TB), TB, 0, stream>>>(mixk, mixkt, (int)E, 32);
    // 3) projections (WMMA GEMMs). Big GEMMs: 16x64 tile/wave (NT=4); mix: NT=2.
    gemm_bf16_kernel<4><<<(1024 / 16) * (4096 / 64) / 8, TB, 0, stream>>>(hsb, wqt, qf, 1024, 4096, 4096);
    gemm_bf16_kernel<4><<<(1024 / 16) * (1024 / 64) / 8, TB, 0, stream>>>(hsb, wkt, kf, 1024, 1024, 4096);
    gemm_bf16_kernel<4><<<(1024 / 16) * (1024 / 64) / 8, TB, 0, stream>>>(hsb, wvt, vf, 1024, 1024, 4096);
    gemm_bf16_kernel<2><<<(1024 / 16) * (32 / 32) / 8, TB, 0, stream>>>(hsb, mixqt, lrq, 1024, 32, 4096);
    gemm_bf16_kernel<2><<<(1024 / 16) * (32 / 32) / 8, TB, 0, stream>>>(hsb, mixkt, lrk, 1024, 32, 4096);
    // 4) RoPE -> head-major bf16
    rope_kernel<<<(int)((S * HD + TB - 1) / TB), TB, 0, stream>>>(qf, qrb, 32);
    rope_kernel<<<(int)((S * KVD + TB - 1) / TB), TB, 0, stream>>>(kf, krb, 8);
    // 5) V transpose -> [KVH][D][S] bf16
    transpose_f2b_kernel<<<(int)((S * KVD + TB - 1) / TB), TB, 0, stream>>>(vf, vtb, (int)S, (int)KVD);
    // 6) positional mix (streams the 1GB q_pos/k_pos tensors, coalesced)
    mixpos_kernel<<<(int)((32 * S * 128 + TB - 1) / TB), TB, 0, stream>>>(lrq, q_pos, pqb);
    mixpos_kernel<<<(int)((32 * S * 128 + TB - 1) / TB), TB, 0, stream>>>(lrk, k_pos, pkb);
    // 7) fused sparse attention
    attn_kernel<<<dim3(1024 / 16, 32), TB, 0, stream>>>(qrb, krb, pqb, pkb, vtb, attnb);
    // 8) output projection -> d_out (f32)
    gemm_bf16_kernel<4><<<(1024 / 16) * (4096 / 64) / 8, TB, 0, stream>>>(attnb, wot, out, 1024, 4096, 4096);
}
